// GraphTransformer_10986526343700
// MI455X (gfx1250) — matrix operand order repaired and checked by reading, not compiled, not run on hardware
//
#include <hip/hip_runtime.h>
#include <hip/hip_bf16.h>
#include <math.h>

// ---------------- problem constants (from reference) ----------------
#define N_NODES 4096
#define N_EDGES 262144
#define F_IN    256
#define F_HID   512
#define F_OUT   128
#define N_HEADS 8
#define HEAD_D  64
#define N_GRAPH 8

typedef __bf16 bf16_t;
typedef __attribute__((ext_vector_type(8)))  bf16_t v8bf;
typedef __attribute__((ext_vector_type(16))) bf16_t v16bf;
typedef __attribute__((ext_vector_type(8)))  float  v8f;

static __device__ __forceinline__ bf16_t f2bf(float f) { return (bf16_t)f; }

static __device__ __forceinline__ v8f wmma_bf16(v16bf a, v16bf b, v8f c) {
  // (neg_a, A, neg_b, B, c_mod, C, reuse_a, reuse_b)
  return __builtin_amdgcn_wmma_f32_16x16x32_bf16(false, a, false, b, (short)0, c,
                                                 false, false);
}

// ---------------- small utility kernels ----------------
__global__ void k_zero_f32(float* __restrict__ p, int n) {
  int i = blockIdx.x * blockDim.x + threadIdx.x;
  if (i < n) p[i] = 0.f;
}

__global__ void k_cast_bf16(const float* __restrict__ in, bf16_t* __restrict__ out, int n) {
  int i = blockIdx.x * blockDim.x + threadIdx.x;
  if (i < n) out[i] = f2bf(in[i]);
}

// W row-major [K][N] (in,out) -> Wt bf16 [N][K] (out,in)
__global__ void k_transpose_bf16(const float* __restrict__ in, bf16_t* __restrict__ out,
                                 int K, int N) {
  int i = blockIdx.x * blockDim.x + threadIdx.x;
  if (i < K * N) {
    int k = i / N, n = i - k * N;
    out[(size_t)n * K + k] = f2bf(in[i]);
  }
}

__global__ void k_degree(const int* __restrict__ dst, float* __restrict__ deg) {
  int e = blockIdx.x * blockDim.x + threadIdx.x;
  if (e < N_EDGES) atomicAdd(&deg[dst[e]], 1.f);
}

__global__ void k_dinv(float* __restrict__ deg) {
  int i = blockIdx.x * blockDim.x + threadIdx.x;
  if (i < N_NODES) deg[i] = rsqrtf(deg[i] + 1.f);   // +1 for self loop; in-place -> dinv
}

// agg = xw * dinv^2 (self-loop contribution; also initializes the accumulator)
__global__ void k_selfloop_init(const float* __restrict__ xw, const float* __restrict__ dinv,
                                float* __restrict__ agg) {
  int i = blockIdx.x * blockDim.x + threadIdx.x;
  if (i < N_NODES * F_HID) {
    int n = i >> 9;                       // / F_HID
    float di = dinv[n];
    agg[i] = xw[i] * di * di;
  }
}

// one thread = 4 features of one edge
__global__ void k_edge_scatter(const int* __restrict__ src, const int* __restrict__ dst,
                               const float* __restrict__ dinv, const float* __restrict__ xw,
                               float* __restrict__ agg) {
  int idx = blockIdx.x * blockDim.x + threadIdx.x;
  int e = idx >> 7;
  if (e >= N_EDGES) return;
  int c = (idx & 127) * 4;
  int s = src[e], d = dst[e];
  float nm = dinv[s] * dinv[d];
  const float4 x = *(const float4*)&xw[(size_t)s * F_HID + c];
  float* ap = &agg[(size_t)d * F_HID + c];
  atomicAdd(ap + 0, x.x * nm);
  atomicAdd(ap + 1, x.y * nm);
  atomicAdd(ap + 2, x.z * nm);
  atomicAdd(ap + 3, x.w * nm);
}

__global__ void k_bias_relu_bf16(const float* __restrict__ agg, const float* __restrict__ b,
                                 bf16_t* __restrict__ hbf) {
  int i = blockIdx.x * blockDim.x + threadIdx.x;
  if (i < N_NODES * F_HID) {
    float v = agg[i] + b[i & 511];
    hbf[i] = f2bf(fmaxf(v, 0.f));
  }
}

// ---------------- WMMA GEMM: C[M,N] = A[M,K] @ Bt[N,K]^T (+bias) ----------------
// MODE 0: C fp32 [M][N] (+bias if non-null)
// MODE 1: QKV split epilogue -> Q [h][n][64] (scaled 1/8), K [h][n][64], Vt [h][64][n]
template <int MODE>
__global__ __launch_bounds__(128)
void k_wmma_gemm(const bf16_t* __restrict__ A, const bf16_t* __restrict__ Bt,
                 const float* __restrict__ bias, float* __restrict__ C,
                 bf16_t* __restrict__ qo, bf16_t* __restrict__ ko, bf16_t* __restrict__ vo,
                 int M, int N, int K) {
  const int lane = threadIdx.x & 31;
  const int wid  = threadIdx.x >> 5;
  const int half = lane >> 4;        // 0 or 1 (wave32 halves)
  const int ln   = lane & 15;
  const int row0 = (blockIdx.y * 4 + wid) * 16;
  const int col0 = blockIdx.x * 64;

  v8f acc[4] = {v8f{}, v8f{}, v8f{}, v8f{}};

  const bf16_t* arow = A + (size_t)(row0 + ln) * K + half * 16;
  for (int k0 = 0; k0 < K; k0 += 32) {
    // A fragment (16x32 bf16): lanes 0-15 K 0..15, lanes 16-31 K 16..31
    v16bf af;
    {
      v8bf lo = *(const v8bf*)(arow + k0);
      v8bf hi = *(const v8bf*)(arow + k0 + 8);
#pragma unroll
      for (int i = 0; i < 8; ++i) { af[i] = lo[i]; af[i + 8] = hi[i]; }
    }
#pragma unroll
    for (int j = 0; j < 4; ++j) {
      // B fragment (32x16): lane = column, lanes 0-15 K 0..15, lanes 16-31 K 16..31
      const bf16_t* bp = Bt + (size_t)(col0 + j * 16 + ln) * K + k0 + half * 16;
      v16bf bfr = *(const v16bf*)bp;
      acc[j] = wmma_bf16(af, bfr, acc[j]);
    }
  }

  const int rbase = row0 + half * 8;   // C layout: VGPR g -> row g (+8 for upper half)
#pragma unroll
  for (int j = 0; j < 4; ++j) {
    int col = col0 + j * 16 + ln;
    float bv = bias ? bias[col] : 0.f;
#pragma unroll
    for (int g = 0; g < 8; ++g) {
      int r = rbase + g;
      float v = acc[j][g] + bv;
      if (MODE == 0) {
        C[(size_t)r * N + col] = v;
      } else {
        int sec = col >> 9, head = (col >> 6) & 7, d = col & 63;
        if (sec == 0)
          qo[((size_t)head * N_NODES + r) * HEAD_D + d] = f2bf(v * 0.125f);
        else if (sec == 1)
          ko[((size_t)head * N_NODES + r) * HEAD_D + d] = f2bf(v);
        else
          vo[((size_t)head * HEAD_D + d) * N_NODES + r] = f2bf(v);
      }
    }
  }
}

// ---------------- flash attention: one wave per (head, 16-query tile) ----------------
// Bc = 64 keys per iteration: WMMA count per key unchanged, but the cross-lane
// softmax reductions (ds_bpermute chains) are amortized over 4 S-subtiles.
__global__ __launch_bounds__(256)
void k_flash_attn(const bf16_t* __restrict__ Q, const bf16_t* __restrict__ Km,
                  const bf16_t* __restrict__ Vt, bf16_t* __restrict__ O) {
  __shared__ bf16_t Pb[8][16][72];   // per-wave 16x64 P tile (padded rows, 16B-aligned)
  const int lane = threadIdx.x & 31;
  const int wid  = threadIdx.x >> 5;
  const int tile = blockIdx.x * 8 + wid;   // 0..2047
  const int head = tile >> 8;
  const int q0   = (tile & 255) * 16;
  const int half = lane >> 4;
  const int ln   = lane & 15;

  // Q fragments for K-dim 0..63 (pre-scaled by 1/8 in the QKV epilogue)
  v16bf qf[2];
  {
    const bf16_t* qr = Q + ((size_t)head * N_NODES + q0 + ln) * HEAD_D + half * 16;
#pragma unroll
    for (int c = 0; c < 2; ++c) {
      v8bf lo = *(const v8bf*)(qr + c * 32);
      v8bf hi = *(const v8bf*)(qr + c * 32 + 8);
#pragma unroll
      for (int i = 0; i < 8; ++i) { qf[c][i] = lo[i]; qf[c][i + 8] = hi[i]; }
    }
  }

  v8f oacc[4] = {v8f{}, v8f{}, v8f{}, v8f{}};
  float mrow[8], lrow[8];
#pragma unroll
  for (int g = 0; g < 8; ++g) { mrow[g] = -1e30f; lrow[g] = 0.f; }

  const bf16_t* kbase = Km + (size_t)head * N_NODES * HEAD_D;   // [key][64]
  const bf16_t* vbase = Vt + (size_t)head * HEAD_D * N_NODES;   // [d][key]

  for (int kt = 0; kt < N_NODES; kt += 64) {
    // --- S = Q @ K^T for 64 keys (4 x 16-col subtiles, K-dim 64 = 2 WMMAs each)
    v8f s[4] = {v8f{}, v8f{}, v8f{}, v8f{}};
#pragma unroll
    for (int sub = 0; sub < 4; ++sub) {
      const bf16_t* kr = kbase + (size_t)(kt + sub * 16 + ln) * HEAD_D + half * 16;
      v16bf b0 = *(const v16bf*)(kr);
      v16bf b1 = *(const v16bf*)(kr + 32);
      s[sub] = wmma_bf16(qf[0], b0, s[sub]);
      s[sub] = wmma_bf16(qf[1], b1, s[sub]);
    }
    // --- online softmax; C-layout row = g + 8*half, col = ln + 16*sub
#pragma unroll
    for (int g = 0; g < 8; ++g) {
      float mc = fmaxf(fmaxf(s[0][g], s[1][g]), fmaxf(s[2][g], s[3][g]));
#pragma unroll
      for (int m = 8; m >= 1; m >>= 1) mc = fmaxf(mc, __shfl_xor(mc, m, 32));
      float mn  = fmaxf(mrow[g], mc);
      float fac = __expf(mrow[g] - mn);
      float p0  = __expf(s[0][g] - mn);
      float p1  = __expf(s[1][g] - mn);
      float p2  = __expf(s[2][g] - mn);
      float p3  = __expf(s[3][g] - mn);
      float ps  = (p0 + p1) + (p2 + p3);
#pragma unroll
      for (int m = 8; m >= 1; m >>= 1) ps += __shfl_xor(ps, m, 32);
      lrow[g] = lrow[g] * fac + ps;
      mrow[g] = mn;
      oacc[0][g] = oacc[0][g] * fac; oacc[1][g] = oacc[1][g] * fac;
      oacc[2][g] = oacc[2][g] * fac; oacc[3][g] = oacc[3][g] * fac;
      int r = g + half * 8;
      Pb[wid][r][ln]      = f2bf(p0);
      Pb[wid][r][ln + 16] = f2bf(p1);
      Pb[wid][r][ln + 32] = f2bf(p2);
      Pb[wid][r][ln + 48] = f2bf(p3);
    }
    // --- relayout P (C-layout in LDS) into A fragments (same wave: DS in-order)
    v16bf pf[2];
#pragma unroll
    for (int c = 0; c < 2; ++c) {
      const bf16_t* pr = &Pb[wid][ln][c * 32 + half * 16];
      v8bf lo = *(const v8bf*)(pr);
      v8bf hi = *(const v8bf*)(pr + 8);
#pragma unroll
      for (int i = 0; i < 8; ++i) { pf[c][i] = lo[i]; pf[c][i + 8] = hi[i]; }
    }
    // --- O += P @ V : B columns = head-dim, K rows = keys (contiguous in Vt)
#pragma unroll
    for (int j = 0; j < 4; ++j) {
      const bf16_t* vr = vbase + (size_t)(j * 16 + ln) * N_NODES + kt + half * 16;
      v16bf vb0 = *(const v16bf*)(vr);
      v16bf vb1 = *(const v16bf*)(vr + 32);
      oacc[j] = wmma_bf16(pf[0], vb0, oacc[j]);
      oacc[j] = wmma_bf16(pf[1], vb1, oacc[j]);
    }
  }

  // finalize: divide by row sums, store bf16 [N][512]
#pragma unroll
  for (int j = 0; j < 4; ++j) {
    int col = head * 64 + j * 16 + ln;
#pragma unroll
    for (int g = 0; g < 8; ++g) {
      int r = q0 + g + half * 8;
      O[(size_t)r * F_HID + col] = f2bf(oacc[j][g] / lrow[g]);
    }
  }
}

// ---------------- pooling + final FC ----------------
__global__ void k_pool_sum(const float* __restrict__ h, const int* __restrict__ batch,
                           float* __restrict__ sums) {
  int idx = blockIdx.x * blockDim.x + threadIdx.x;
  int n = idx >> 7;
  if (n >= N_NODES) return;
  int c = (idx & 127) * 4;
  int g = batch[n];
  const float4 v = *(const float4*)&h[(size_t)n * F_HID + c];
  float* sp = &sums[(size_t)g * F_HID + c];
  atomicAdd(sp + 0, v.x); atomicAdd(sp + 1, v.y);
  atomicAdd(sp + 2, v.z); atomicAdd(sp + 3, v.w);
}

__global__ void k_pool_cnt(const int* __restrict__ batch, float* __restrict__ cnt) {
  int n = blockIdx.x * blockDim.x + threadIdx.x;
  if (n < N_NODES) atomicAdd(&cnt[batch[n]], 1.f);
}

__global__ void k_final_fc(const float* __restrict__ sums, const float* __restrict__ cnt,
                           const float* __restrict__ fcw, const float* __restrict__ fcb,
                           float* __restrict__ out) {
  int idx = blockIdx.x * blockDim.x + threadIdx.x;
  if (idx >= N_GRAPH * F_OUT) return;
  int g = idx / F_OUT, o = idx - g * F_OUT;
  float acc = 0.f;
  for (int k2 = 0; k2 < F_HID; ++k2) acc += sums[g * F_HID + k2] * fcw[k2 * F_OUT + o];
  out[idx] = acc / fmaxf(cnt[g], 1.f) + fcb[o];
}

// ---------------- host side ----------------
static inline size_t alignup(size_t x) { return (x + 255) & ~(size_t)255; }

extern "C" void kernel_launch(void* const* d_in, const int* in_sizes, int n_in,
                              void* d_out, int out_size, void* d_ws, size_t ws_size,
                              hipStream_t stream) {
  (void)in_sizes; (void)n_in; (void)out_size; (void)ws_size;
  const float* x     = (const float*)d_in[0];
  const int*   eidx  = (const int*)d_in[1];
  const int*   batch = (const int*)d_in[2];
  const float* W1    = (const float*)d_in[3];
  const float* b1    = (const float*)d_in[4];
  const float* W2    = (const float*)d_in[5];
  const float* b2    = (const float*)d_in[6];
  const float* in_w  = (const float*)d_in[7];
  const float* in_b  = (const float*)d_in[8];
  const float* out_w = (const float*)d_in[9];
  const float* out_b = (const float*)d_in[10];
  const float* fc_w  = (const float*)d_in[11];
  const float* fc_b  = (const float*)d_in[12];
  const int* src = eidx;
  const int* dst = eidx + N_EDGES;
  float* out = (float*)d_out;

  // ---- carve workspace ----
  char* p = (char*)d_ws; size_t off = 0;
  auto carve = [&](size_t bytes) { char* r = p + off; off = alignup(off + bytes); return r; };
  bf16_t* Xbf   = (bf16_t*)carve((size_t)N_NODES * F_IN * 2);
  bf16_t* W1t   = (bf16_t*)carve((size_t)F_HID * F_IN * 2);
  bf16_t* W2t   = (bf16_t*)carve((size_t)F_HID * F_HID * 2);
  bf16_t* INWt  = (bf16_t*)carve((size_t)3 * F_HID * F_HID * 2);
  bf16_t* OUTWt = (bf16_t*)carve((size_t)F_HID * F_HID * 2);
  float*  dinv  = (float*)carve((size_t)N_NODES * 4);
  float*  XW    = (float*)carve((size_t)N_NODES * F_HID * 4);
  float*  AGG   = (float*)carve((size_t)N_NODES * F_HID * 4);
  bf16_t* Hbf   = (bf16_t*)carve((size_t)N_NODES * F_HID * 2);   // h1bf -> h2bf -> Obf
  bf16_t* Qbf   = (bf16_t*)carve((size_t)N_HEADS * N_NODES * HEAD_D * 2);
  bf16_t* Kbf   = (bf16_t*)carve((size_t)N_HEADS * N_NODES * HEAD_D * 2);
  bf16_t* Vtbf  = (bf16_t*)carve((size_t)N_HEADS * HEAD_D * N_NODES * 2);
  float*  H3    = (float*)carve((size_t)N_NODES * F_HID * 4);
  float*  SUMS  = (float*)carve((size_t)N_GRAPH * F_HID * 4);
  float*  CNT   = (float*)carve((size_t)N_GRAPH * 4);

  const int T = 256;
  auto blocks = [](int n, int t) { return (n + t - 1) / t; };

  // ---- weight / input conversion ----
  k_cast_bf16<<<blocks(N_NODES * F_IN, T), T, 0, stream>>>(x, Xbf, N_NODES * F_IN);
  k_transpose_bf16<<<blocks(F_IN * F_HID, T), T, 0, stream>>>(W1, W1t, F_IN, F_HID);
  k_transpose_bf16<<<blocks(F_HID * F_HID, T), T, 0, stream>>>(W2, W2t, F_HID, F_HID);
  k_cast_bf16<<<blocks(3 * F_HID * F_HID, T), T, 0, stream>>>(in_w, INWt, 3 * F_HID * F_HID);
  k_cast_bf16<<<blocks(F_HID * F_HID, T), T, 0, stream>>>(out_w, OUTWt, F_HID * F_HID);

  // ---- degree / norm ----
  k_zero_f32<<<blocks(N_NODES, T), T, 0, stream>>>(dinv, N_NODES);
  k_degree<<<blocks(N_EDGES, T), T, 0, stream>>>(dst, dinv);
  k_dinv<<<blocks(N_NODES, T), T, 0, stream>>>(dinv);

  dim3 gblk(128);
  // ---- GCN layer 1 ----
  k_wmma_gemm<0><<<dim3(F_HID / 64, N_NODES / 64), gblk, 0, stream>>>(
      Xbf, W1t, nullptr, XW, nullptr, nullptr, nullptr, N_NODES, F_HID, F_IN);
  k_selfloop_init<<<blocks(N_NODES * F_HID, T), T, 0, stream>>>(XW, dinv, AGG);
  k_edge_scatter<<<blocks(N_EDGES * 128, T), T, 0, stream>>>(src, dst, dinv, XW, AGG);
  k_bias_relu_bf16<<<blocks(N_NODES * F_HID, T), T, 0, stream>>>(AGG, b1, Hbf);

  // ---- GCN layer 2 ----
  k_wmma_gemm<0><<<dim3(F_HID / 64, N_NODES / 64), gblk, 0, stream>>>(
      Hbf, W2t, nullptr, XW, nullptr, nullptr, nullptr, N_NODES, F_HID, F_HID);
  k_selfloop_init<<<blocks(N_NODES * F_HID, T), T, 0, stream>>>(XW, dinv, AGG);
  k_edge_scatter<<<blocks(N_EDGES * 128, T), T, 0, stream>>>(src, dst, dinv, XW, AGG);
  k_bias_relu_bf16<<<blocks(N_NODES * F_HID, T), T, 0, stream>>>(AGG, b2, Hbf);

  // ---- QKV projection (writes Q scaled, K, V-transposed) ----
  k_wmma_gemm<1><<<dim3(3 * F_HID / 64, N_NODES / 64), gblk, 0, stream>>>(
      Hbf, INWt, in_b, nullptr, Qbf, Kbf, Vtbf, N_NODES, 3 * F_HID, F_HID);

  // ---- flash attention (overwrites Hbf with O, bf16) ----
  k_flash_attn<<<(N_HEADS * (N_NODES / 16)) / 8, 256, 0, stream>>>(Qbf, Kbf, Vtbf, Hbf);

  // ---- output projection ----
  k_wmma_gemm<0><<<dim3(F_HID / 64, N_NODES / 64), gblk, 0, stream>>>(
      Hbf, OUTWt, out_b, H3, nullptr, nullptr, nullptr, N_NODES, F_HID, F_HID);

  // ---- mean pool + final FC ----
  k_zero_f32<<<blocks(N_GRAPH * F_HID, T), T, 0, stream>>>(SUMS, N_GRAPH * F_HID);
  k_zero_f32<<<1, T, 0, stream>>>(CNT, N_GRAPH);
  k_pool_sum<<<blocks(N_NODES * 128, T), T, 0, stream>>>(H3, batch, SUMS);
  k_pool_cnt<<<blocks(N_NODES, T), T, 0, stream>>>(batch, CNT);
  k_final_fc<<<blocks(N_GRAPH * F_OUT, T), T, 0, stream>>>(SUMS, CNT, fc_w, fc_b, out);
}
